// Block_12945031430220
// MI455X (gfx1250) — matrix operand order repaired and checked by reading, not compiled, not run on hardware
//
#include <hip/hip_runtime.h>
#include <hip/hip_bf16.h>
#include <math.h>

typedef _Float16 v16h __attribute__((ext_vector_type(16)));
typedef _Float16 v8h  __attribute__((ext_vector_type(8)));
typedef float    v8f  __attribute__((ext_vector_type(8)));

#define BATCH    32
#define NTOK     385
#define NPAD     416      // 26*16 = 13*32
#define DIM      512
#define QKVD     1536
#define NHEAD    8
#define HD       64
#define MLPH     2048
#define AGENT_N  64
#define MAP_N    192
#define STATIC_N 128
#define KA       45       // ceil(64*0.7)
#define KM       135      // ceil(192*0.7)
#define KS       90       // ceil(128*0.7)
#define RN1      271      // 1+45+135+90
#define PN       384      // patch-attn side

// ---------------- fragment loaders (wave32 WMMA, same contiguous k order in A and B) ----------
// A (16x32 f16): lane L holds row M=L&15, K = (L>>4)*16 + e
__device__ inline v16h fragA_ld(const _Float16* __restrict__ p, int kbase) {
  v8h lo = *reinterpret_cast<const v8h*>(p + 2 * kbase);      // K = 2*kbase..+7
  v8h hi = *reinterpret_cast<const v8h*>(p + 2 * kbase + 8);  // K = 2*kbase+8..+15
  v16h r;
#pragma unroll
  for (int i = 0; i < 8; ++i) { r[i] = lo[i]; r[i + 8] = hi[i]; }
  return r;
}
// B (32x16 f16): lane L holds col N=L&15, K = (L>>4)*16 + e  (contiguous 16 f16)
__device__ inline v16h fragB_ld(const _Float16* __restrict__ p) {
  return *reinterpret_cast<const v16h*>(p);
}

__device__ inline v8f wmma_f16(v16h a, v16h b, v8f c) {
  return __builtin_amdgcn_wmma_f32_16x16x32_f16(false, a, false, b, (short)0, c,
                                                false, false);
}

// ---------------- generic f32->f16 convert --------------------------------
__global__ void f32_to_f16_kernel(const float* __restrict__ in,
                                  _Float16* __restrict__ out, int n) {
  int i = blockIdx.x * blockDim.x + threadIdx.x;
  if (i < n) out[i] = (_Float16)in[i];
}

// ---------------- layernorm -> f16 (row = 512) -----------------------------
__global__ __launch_bounds__(128) void ln_f16_kernel(
    const float* __restrict__ x, const float* __restrict__ g,
    const float* __restrict__ be, _Float16* __restrict__ out, int rows) {
  int row = blockIdx.x;
  if (row >= rows) return;
  const float* xp = x + (size_t)row * DIM;
  int tid = threadIdx.x;
  float v[4], s = 0.f, s2 = 0.f;
#pragma unroll
  for (int i = 0; i < 4; ++i) {
    v[i] = xp[tid + 128 * i];
    s += v[i]; s2 += v[i] * v[i];
  }
  __shared__ float rs[128], rq[128];
  rs[tid] = s; rq[tid] = s2;
  __syncthreads();
  for (int o = 64; o >= 1; o >>= 1) {
    if (tid < o) { rs[tid] += rs[tid + o]; rq[tid] += rq[tid + o]; }
    __syncthreads();
  }
  float mean = rs[0] * (1.f / DIM);
  float var  = rq[0] * (1.f / DIM) - mean * mean;
  float inv  = rsqrtf(var + 1e-5f);
  _Float16* op = out + (size_t)row * DIM;
#pragma unroll
  for (int i = 0; i < 4; ++i) {
    int c = tid + 128 * i;
    op[c] = (_Float16)((v[i] - mean) * inv * g[c] + be[c]);
  }
}

// ---------------- WMMA GEMM: C = A(MxK,f16) * W(NxK,f16)^T ------------------
// epi: 0 = store f16 ; 1 = +bias +res -> f32 ; 2 = +bias, exact GELU -> f16
__global__ __launch_bounds__(32) void gemm_wmma_kernel(
    const _Float16* __restrict__ A, const _Float16* __restrict__ W,
    float* __restrict__ out32, _Float16* __restrict__ out16,
    const float* __restrict__ bias, const float* __restrict__ res,
    int M, int N, int K, int epi) {
  const int lane    = threadIdx.x & 31;
  const int nlo     = lane & 15;
  const int kbase   = (lane >> 4) * 8;
  const int khalf16 = (lane >> 4) * 16;
  const int mhi     = (lane >> 4) * 8;
  const int m0 = blockIdx.x * 16;
  const int n0 = blockIdx.y * 64;
  int arow = m0 + nlo; if (arow > M - 1) arow = M - 1;
  const _Float16* Ap = A + (size_t)arow * K;
  v8f acc[4];
#pragma unroll
  for (int t = 0; t < 4; ++t)
#pragma unroll
    for (int i = 0; i < 8; ++i) acc[t][i] = 0.f;
  for (int k0 = 0; k0 < K; k0 += 32) {
    v16h a = fragA_ld(Ap + k0, kbase);
#pragma unroll
    for (int t = 0; t < 4; ++t) {
      const _Float16* Wp = W + (size_t)(n0 + t * 16 + nlo) * K + k0 + khalf16;
      acc[t] = wmma_f16(a, fragB_ld(Wp), acc[t]);
    }
  }
#pragma unroll
  for (int t = 0; t < 4; ++t) {
    int col = n0 + t * 16 + nlo;
    float bv = (epi != 0) ? bias[col] : 0.f;
#pragma unroll
    for (int r = 0; r < 8; ++r) {
      int m = m0 + mhi + r;
      if (m >= M) continue;
      size_t o = (size_t)m * N + col;
      float v = acc[t][r];
      if (epi == 0) {
        out16[o] = (_Float16)v;
      } else if (epi == 1) {
        out32[o] = v + bv + res[o];
      } else {
        float u = v + bv;
        out16[o] = (_Float16)(0.5f * u * (1.f + erff(u * 0.70710678118654752440f)));
      }
    }
  }
}

// ---------------- fused attention (per batch x 16-row tile) -----------------
__global__ __launch_bounds__(32) void attn_kernel(
    const _Float16* __restrict__ qkv,     // [B,NTOK,3*DIM]: q|k|v
    const float* __restrict__ amask,      // [B,NTOK,NTOK]
    _Float16* __restrict__ xatt_h,        // [B,NTOK,DIM] (f16)
    float* __restrict__ cls,              // [B,PN]
    float* __restrict__ patch) {          // [B,PN,PN]
  __shared__ float S[16][NPAD];
  __shared__ float PAcc[16][PN];
  __shared__ float CAcc[PN];
  const int b  = blockIdx.x;
  const int rt = blockIdx.y;              // 0..24
  const int lane    = threadIdx.x & 31;
  const int nlo     = lane & 15;
  const int kbase   = (lane >> 4) * 8;
  const int khalf16 = (lane >> 4) * 16;
  const int mhi     = (lane >> 4) * 8;
  for (int i = lane; i < 16 * PN; i += 32) (&PAcc[0][0])[i] = 0.f;
  if (rt == 0) for (int i = lane; i < PN; i += 32) CAcc[i] = 0.f;
  __syncthreads();
  const int rowA = rt * 16 + nlo;
  const int arow = rowA < NTOK ? rowA : NTOK - 1;
  const float scale = 0.125f;             // 64^-0.5
  for (int h = 0; h < NHEAD; ++h) {
    // ---- scores = (q . k^T) * scale -> LDS ----
    const _Float16* qp = qkv + (size_t)(b * NTOK + arow) * QKVD + h * HD;
    v16h q0 = fragA_ld(qp, kbase);
    v16h q1 = fragA_ld(qp + 32, kbase);
    for (int ct = 0; ct < NPAD / 16; ++ct) {
      int col = ct * 16 + nlo;
      int acol = col < NTOK ? col : NTOK - 1;
      const _Float16* kp =
          qkv + (size_t)(b * NTOK + acol) * QKVD + DIM + h * HD + khalf16;
      v8f sacc;
#pragma unroll
      for (int i = 0; i < 8; ++i) sacc[i] = 0.f;
      sacc = wmma_f16(q0, fragB_ld(kp), sacc);
      sacc = wmma_f16(q1, fragB_ld(kp + 32), sacc);
#pragma unroll
      for (int r = 0; r < 8; ++r) S[mhi + r][ct * 16 + nlo] = sacc[r] * scale;
    }
    __syncthreads();
    // ---- softmax (reference variant) on lanes 0..15 ----
    if (lane < 16) {
      int grow = rt * 16 + lane;
      if (grow < NTOK) {
        float mx = -3.4e38f;
        for (int c = 0; c < NTOK; ++c) mx = fmaxf(mx, S[lane][c]);
        const float* mrow = amask + ((size_t)b * NTOK + grow) * NTOK;
        float sum = 0.f;
        for (int c = 0; c < NTOK; ++c) {
          float p = __expf(S[lane][c] - mx) * mrow[c];
          S[lane][c] = p; sum += p;
        }
        float inv = 1.f / (sum + 1e-6f);
        float addc = 1e-6f / (float)NTOK;
        for (int c = 0; c < NTOK; ++c) S[lane][c] = (S[lane][c] + addc) * inv;
        for (int c = NTOK; c < NPAD; ++c) S[lane][c] = 0.f;
        if (grow == 0) {
          for (int c = 1; c < NTOK; ++c) CAcc[c - 1] += S[lane][c] * 0.125f;
        } else {
          for (int c = 1; c < NTOK; ++c) PAcc[lane][c - 1] += S[lane][c] * 0.125f;
        }
      } else {
        for (int c = 0; c < NPAD; ++c) S[lane][c] = 0.f;
      }
    }
    __syncthreads();
    // ---- out = P . V  (A frag gathered from LDS, f32->f16 convert) ----
    v8f oacc[4];
#pragma unroll
    for (int t = 0; t < 4; ++t)
#pragma unroll
      for (int i = 0; i < 8; ++i) oacc[t][i] = 0.f;
    for (int kc = 0; kc < NPAD / 32; ++kc) {
      v16h a;
#pragma unroll
      for (int e = 0; e < 16; ++e) {
        int c = kc * 32 + khalf16 + e;
        a[e] = (_Float16)S[nlo][c];
      }
#pragma unroll
      for (int dt = 0; dt < 4; ++dt) {
        v16h bf;
#pragma unroll
        for (int e = 0; e < 16; ++e) {
          int c = kc * 32 + khalf16 + e;
          if (c > NTOK - 1) c = NTOK - 1;           // P is 0 there anyway
          bf[e] = qkv[(size_t)(b * NTOK + c) * QKVD + 2 * DIM + h * HD +
                      dt * 16 + nlo];
        }
        oacc[dt] = wmma_f16(a, bf, oacc[dt]);
      }
    }
#pragma unroll
    for (int dt = 0; dt < 4; ++dt)
#pragma unroll
      for (int r = 0; r < 8; ++r) {
        int grow = rt * 16 + mhi + r;
        if (grow < NTOK)
          xatt_h[(size_t)(b * NTOK + grow) * DIM + h * HD + dt * 16 + nlo] =
              (_Float16)oacc[dt][r];
      }
    __syncthreads();
  }
  // ---- flush per-tile cls/patch accumulators ----
  for (int m = 0; m < 16; ++m) {
    int grow = rt * 16 + m;
    if (grow >= 1 && grow < NTOK)
      for (int c = lane; c < PN; c += 32)
        patch[((size_t)b * PN + (grow - 1)) * PN + c] = PAcc[m][c];
  }
  if (rt == 0)
    for (int c = lane; c < PN; c += 32) cls[(size_t)b * PN + c] = CAcc[c];
}

// ---------------- per-segment top-k (stable, matches lax.top_k ties) --------
__global__ __launch_bounds__(64) void topk_kernel(
    const float* __restrict__ cls, int* __restrict__ ia, int* __restrict__ im,
    int* __restrict__ is) {
  const int b = blockIdx.x;
  const int seg = blockIdx.y;
  int off, n, k; int* out;
  if (seg == 0)      { off = 0;               n = AGENT_N;  k = KA; out = ia + b * KA; }
  else if (seg == 1) { off = AGENT_N;         n = MAP_N;    k = KM; out = im + b * KM; }
  else               { off = AGENT_N + MAP_N; n = STATIC_N; k = KS; out = is + b * KS; }
  __shared__ float vals[MAP_N];
  __shared__ int taken[MAP_N];
  __shared__ float bv[64];
  __shared__ int bi[64];
  int tid = threadIdx.x;
  for (int i = tid; i < n; i += 64) {
    vals[i] = cls[(size_t)b * PN + off + i];
    taken[i] = 0;
  }
  __syncthreads();
  for (int j = 0; j < k; ++j) {
    float v = -3.4e38f; int id = n;
    for (int i = tid; i < n; i += 64)
      if (!taken[i] && vals[i] > v) { v = vals[i]; id = i; }  // keeps lowest i
    bv[tid] = v; bi[tid] = id;
    __syncthreads();
    if (tid == 0) {
      float best = -3.4e38f; int bid = n;
      for (int t = 0; t < 64; ++t)
        if (bi[t] < n && (bv[t] > best || (bv[t] == best && bi[t] < bid))) {
          best = bv[t]; bid = bi[t];
        }
      taken[bid] = 1; out[j] = bid;
    }
    __syncthreads();
  }
}

// ---------------- gather pruned rows ---------------------------------------
__global__ __launch_bounds__(128) void gather_kernel(
    const float* __restrict__ x1, const int* __restrict__ ia,
    const int* __restrict__ im, const int* __restrict__ is,
    float* __restrict__ xo) {
  int blk = blockIdx.x;            // B*RN1
  int b = blk / RN1, o = blk % RN1;
  int src;
  if (o == 0)            src = 0;
  else if (o <= KA)      src = 1 + ia[b * KA + (o - 1)];
  else if (o <= KA + KM) src = 1 + AGENT_N + im[b * KM + (o - 1 - KA)];
  else                   src = 1 + AGENT_N + MAP_N + is[b * KS + (o - 1 - KA - KM)];
  const float* sp = x1 + ((size_t)b * NTOK + src) * DIM;
  float* dp = xo + ((size_t)b * RN1 + o) * DIM;
  for (int i = threadIdx.x; i < DIM; i += 128) dp[i] = sp[i];
}

// ---------------- per-row kth-largest threshold ----------------------------
__global__ __launch_bounds__(128) void thresh_kernel(
    const float* __restrict__ patch, const int* __restrict__ ia,
    const int* __restrict__ im, const int* __restrict__ is,
    float* __restrict__ thr) {
  const int b = blockIdx.x;
  const int r = blockIdx.y;        // 0..269
  int off, L, kk, i; const int* idx;
  if (r < KA)           { off = 0;               L = KA; kk = KA; idx = ia + b * KA; i = r; }
  else if (r < KA + KM) { off = AGENT_N;         L = KM; kk = KM; idx = im + b * KM; i = r - KA; }
  else                  { off = AGENT_N + MAP_N; L = KS; kk = KS; idx = is + b * KS; i = r - KA - KM; }
  __shared__ float pa[KM];
  int tid = threadIdx.x;
  const float* prow = patch + ((size_t)b * PN + off + idx[i]) * PN + off;
  for (int j = tid; j < L; j += 128) pa[j] = prow[idx[j]];
  __syncthreads();
  // kth largest = max{v in row : count(row >= v) >= kk}
  float best = -3.4e38f;
  for (int c = tid; c < L; c += 128) {
    float v = pa[c]; int cnt = 0;
    for (int j = 0; j < L; ++j) cnt += (pa[j] >= v) ? 1 : 0;
    if (cnt >= kk && v > best) best = v;
  }
  __shared__ float red[128];
  red[tid] = best;
  __syncthreads();
  for (int o = 64; o >= 1; o >>= 1) {
    if (tid < o) red[tid] = fmaxf(red[tid], red[tid + o]);
    __syncthreads();
  }
  if (tid == 0) thr[(size_t)b * 270 + r] = red[0];
}

// ---------------- build new 271x271 mask -----------------------------------
__global__ void mask_kernel(const float* __restrict__ patch,
                            const int* __restrict__ ia, const int* __restrict__ im,
                            const int* __restrict__ is, const float* __restrict__ thr,
                            float* __restrict__ mout) {
  int t = blockIdx.x * blockDim.x + threadIdx.x;
  const int total = BATCH * RN1 * RN1;
  if (t >= total) return;
  int b = t / (RN1 * RN1);
  int rem = t % (RN1 * RN1);
  int r = rem / RN1, c = rem % RN1;
  float v = 1.f;
  if (r >= 1 && c >= 1) {
    int off = -1, s = 0; const int* idx = nullptr;
    if (r <= KA && c <= KA)                       { off = 0;               s = 1;           idx = ia + b * KA; }
    else if (r > KA && r <= KA + KM &&
             c > KA && c <= KA + KM)              { off = AGENT_N;         s = KA + 1;      idx = im + b * KM; }
    else if (r > KA + KM && c > KA + KM)          { off = AGENT_N + MAP_N; s = KA + KM + 1; idx = is + b * KS; }
    if (off >= 0) {
      int i = r - s, j = c - s;
      float pav = patch[((size_t)b * PN + off + idx[i]) * PN + off + idx[j]];
      v = (pav >= thr[(size_t)b * 270 + (r - 1)]) ? 1.f : 0.f;
    }
  }
  mout[t] = v;
}

// ============================ host orchestration ============================
extern "C" void kernel_launch(void* const* d_in, const int* in_sizes, int n_in,
                              void* d_out, int out_size, void* d_ws, size_t ws_size,
                              hipStream_t stream) {
  const float* x     = (const float*)d_in[0];
  const float* amask = (const float*)d_in[1];
  const float* Wqkv  = (const float*)d_in[5];
  const float* Wproj = (const float*)d_in[6];
  const float* bproj = (const float*)d_in[7];
  const float* g1    = (const float*)d_in[8];
  const float* b1    = (const float*)d_in[9];
  const float* g2    = (const float*)d_in[10];
  const float* b2    = (const float*)d_in[11];
  const float* Wfc1  = (const float*)d_in[12];
  const float* bfc1  = (const float*)d_in[13];
  const float* Wfc2  = (const float*)d_in[14];
  const float* bfc2  = (const float*)d_in[15];

  const int MROWS  = BATCH * NTOK;   // 12320 (div by 16)
  const int MROWS2 = BATCH * RN1;    // 8672  (div by 16)

  // workspace bump allocator
  char* ws = (char*)d_ws;
  size_t off = 0;
  auto alloc = [&](size_t bytes) -> char* {
    char* p = ws + off;
    off = (off + bytes + 255) & ~(size_t)255;
    return p;
  };
  _Float16* wqkv_h  = (_Float16*)alloc((size_t)QKVD * DIM * 2);
  _Float16* wproj_h = (_Float16*)alloc((size_t)DIM * DIM * 2);
  _Float16* wfc1_h  = (_Float16*)alloc((size_t)MLPH * DIM * 2);
  _Float16* wfc2_h  = (_Float16*)alloc((size_t)DIM * MLPH * 2);
  _Float16* xn_h    = (_Float16*)alloc((size_t)MROWS * DIM * 2);   // reused for LN2
  _Float16* qkv_h   = (_Float16*)alloc((size_t)MROWS * QKVD * 2);  // reused for MLP hidden
  _Float16* xatt_h  = (_Float16*)alloc((size_t)MROWS * DIM * 2);
  float*    x1      = (float*)   alloc((size_t)MROWS * DIM * 4);
  float*    cls     = (float*)   alloc((size_t)BATCH * PN * 4);
  float*    patch   = (float*)   alloc((size_t)BATCH * PN * PN * 4);
  float*    thr     = (float*)   alloc((size_t)BATCH * 270 * 4);
  _Float16* h_h     = qkv_h;  // 8672*2048 f16 fits in qkv_h region

  // output carving: x | idx_a | idx_m | idx_s | new_mask
  float* o  = (float*)d_out;
  float* xo = o;
  int* ia = (int*)(o + (size_t)BATCH * RN1 * DIM);
  int* im = ia + BATCH * KA;
  int* is = im + BATCH * KM;
  float* mout = (float*)(is + BATCH * KS);

  // 1) weights -> f16
  {
    int n;
    n = QKVD * DIM; f32_to_f16_kernel<<<(n + 255) / 256, 256, 0, stream>>>(Wqkv, wqkv_h, n);
    n = DIM * DIM;  f32_to_f16_kernel<<<(n + 255) / 256, 256, 0, stream>>>(Wproj, wproj_h, n);
    n = MLPH * DIM; f32_to_f16_kernel<<<(n + 255) / 256, 256, 0, stream>>>(Wfc1, wfc1_h, n);
    n = DIM * MLPH; f32_to_f16_kernel<<<(n + 255) / 256, 256, 0, stream>>>(Wfc2, wfc2_h, n);
  }
  // 2) LN1 -> f16
  ln_f16_kernel<<<MROWS, 128, 0, stream>>>(x, g1, b1, xn_h, MROWS);
  // 3) QKV GEMM (f16 out)
  gemm_wmma_kernel<<<dim3(MROWS / 16, QKVD / 64), 32, 0, stream>>>(
      xn_h, wqkv_h, nullptr, qkv_h, nullptr, nullptr, MROWS, QKVD, DIM, 0);
  // 4) fused attention (writes xatt_h f16, cls, patch)
  attn_kernel<<<dim3(BATCH, 25), 32, 0, stream>>>(qkv_h, amask, xatt_h, cls, patch);
  // 5) proj GEMM + bias + residual(x) -> x1 (f32)
  gemm_wmma_kernel<<<dim3(MROWS / 16, DIM / 64), 32, 0, stream>>>(
      xatt_h, wproj_h, x1, nullptr, bproj, x, MROWS, DIM, DIM, 1);
  // 6) top-k per segment -> idx outputs
  topk_kernel<<<dim3(BATCH, 3), 64, 0, stream>>>(cls, ia, im, is);
  // 7) gather pruned tokens -> output x region
  gather_kernel<<<BATCH * RN1, 128, 0, stream>>>(x1, ia, im, is, xo);
  // 8) per-row kth-largest thresholds
  thresh_kernel<<<dim3(BATCH, 270), 128, 0, stream>>>(patch, ia, im, is, thr);
  // 9) new mask
  {
    int n = BATCH * RN1 * RN1;
    mask_kernel<<<(n + 255) / 256, 256, 0, stream>>>(patch, ia, im, is, thr, mout);
  }
  // 10) LN2 on pruned x -> f16
  ln_f16_kernel<<<MROWS2, 128, 0, stream>>>(xo, g2, b2, xn_h, MROWS2);
  // 11) fc1 GEMM + bias + exact GELU -> f16 hidden
  gemm_wmma_kernel<<<dim3(MROWS2 / 16, MLPH / 64), 32, 0, stream>>>(
      xn_h, wfc1_h, nullptr, h_h, bfc1, nullptr, MROWS2, MLPH, DIM, 2);
  // 12) fc2 GEMM + bias + residual(xo) -> xo (f32)
  gemm_wmma_kernel<<<dim3(MROWS2 / 16, DIM / 64), 32, 0, stream>>>(
      h_h, wfc2_h, xo, nullptr, bfc2, xo, MROWS2, DIM, MLPH, 1);

  (void)in_sizes; (void)n_in; (void)out_size; (void)ws_size;
}
